// MACEDiffusionAdapted_60851096650035
// MI455X (gfx1250) — matrix operand order repaired and checked by reading, not compiled, not run on hardware
//
#include <hip/hip_runtime.h>
#include <math.h>

// Problem constants (match reference config)
#define BB 4
#define NN 128
#define FF 128
#define LLAYERS 2
#define RMAXF 10.0f

typedef __attribute__((ext_vector_type(16))) _Float16 v16h;
typedef __attribute__((ext_vector_type(8)))  float    v8f;

// ---------------------------------------------------------------------------
// helpers
// ---------------------------------------------------------------------------
__device__ __forceinline__ int ell_of(int k) {
  return (k == 0) ? 0 : (k < 4) ? 1 : (k < 9) ? 2 : 3;
}

__device__ __forceinline__ float silu_f(float x) { return x / (1.f + expf(-x)); }

// real spherical harmonics up to l=3 (component normalization), unit input
__device__ __forceinline__ float sh_comp(int k, float x, float y, float z) {
  const float s3    = 1.7320508075688772f;
  const float s5    = 2.2360679774997896f;
  const float s15   = 3.8729833462074170f;
  const float s35_8 = 2.0916500663351889f;   // sqrt(35/8)
  const float s105  = 10.246950765959598f;
  const float s21_8 = 1.6201851746019651f;   // sqrt(21/8)
  const float s7    = 2.6457513110645907f;
  switch (k) {
    case 0:  return 1.f;
    case 1:  return s3 * x;
    case 2:  return s3 * y;
    case 3:  return s3 * z;
    case 4:  return s15 * x * y;
    case 5:  return s15 * y * z;
    case 6:  return 0.5f * s5 * (3.f * z * z - 1.f);
    case 7:  return s15 * x * z;
    case 8:  return 0.5f * s15 * (x * x - y * y);
    case 9:  return s35_8 * y * (3.f * x * x - y * y);
    case 10: return s105 * x * y * z;
    case 11: return s21_8 * y * (5.f * z * z - 1.f);
    case 12: return 0.5f * s7 * z * (5.f * z * z - 3.f);
    case 13: return s21_8 * x * (5.f * z * z - 1.f);
    default: break;
  }
  if (k == 14) return 0.5f * s105 * z * (x * x - y * y);
  return s35_8 * x * (x * x - 3.f * y * y);  // k == 15
}

// A-operand loader with on-the-fly f32->f16 (used only in the tiny node GEMM).
// A is 16x32 (MxK), row-major with leading dim ld; lane = tid in wave.
// CDNA5 16-bit A layout: lanes0-15 hold K 0-7,16-23 ; lanes16-31 K 8-15,24-31.
__device__ __forceinline__ v16h wmma_load_a(const float* src, int ld, int lane) {
  int m  = lane & 15;
  int hi = (lane >> 4) & 1;
  v16h a;
#pragma unroll
  for (int j = 0; j < 16; ++j) {
    int k = ((j < 8) ? j : (j + 8)) + hi * 8;
    a[j] = (_Float16)src[m * ld + k];
  }
  return a;
}

__device__ __forceinline__ v8f wmma_f16f32(v16h a, v16h b, v8f c) {
  return __builtin_amdgcn_wmma_f32_16x16x32_f16(false, a, false, b, (short)0, c,
                                                false, false);
}

// ---------------------------------------------------------------------------
// Weight pre-swizzle: pack f32 weights into f16 in the exact per-lane v16h
// B-operand register layout (element j of lane holds K = kchunk*32+(j<8?j:j+8)+8*hi,
// N = ntile*16 + (lane&15)), so the GEMM loop does one contiguous 32B load
// per operand instead of 16 scattered dword loads + converts.
// ---------------------------------------------------------------------------
__global__ __launch_bounds__(256) void swizzle_wr2_kernel(
    const float* __restrict__ Wr2, _Float16* __restrict__ out) {
  int o = blockIdx.x * 256 + threadIdx.x;  // L*64*512 = 65536 halves
  int j    = o & 15;
  int lane = (o >> 4) & 31;
  int nt   = (o >> 9) & 31;
  int kc   = (o >> 14) & 1;
  int i    = o >> 15;
  int k = kc * 32 + ((j < 8) ? j : (j + 8)) + 8 * (lane >> 4);
  int n = nt * 16 + (lane & 15);
  out[o] = (_Float16)Wr2[(size_t)i * 64 * 512 + (size_t)k * 512 + n];
}

__global__ __launch_bounds__(256) void swizzle_wnode_kernel(
    const float* __restrict__ Wn, _Float16* __restrict__ out) {
  int o = blockIdx.x * 256 + threadIdx.x;  // L*128*128 = 32768 halves
  int j    = o & 15;
  int lane = (o >> 4) & 31;
  int nt   = (o >> 9) & 7;
  int kc   = (o >> 12) & 3;
  int i    = o >> 14;
  int k = kc * 32 + ((j < 8) ? j : (j + 8)) + 8 * (lane >> 4);
  int n = nt * 16 + (lane & 15);
  out[o] = (_Float16)Wn[(size_t)i * FF * FF + (size_t)k * FF + n];
}

// ---------------------------------------------------------------------------
// K0: embedding  scal = onehot(species) @ W_embed + te @ W_embed[S:] + b
//     also zero vec state and copy positions into the ping buffer.
// ---------------------------------------------------------------------------
__global__ __launch_bounds__(128) void embed_kernel(
    const float* __restrict__ pos, const int* __restrict__ nf,
    const float* __restrict__ gf, const float* __restrict__ W_embed,
    const float* __restrict__ b_embed, float* __restrict__ pos_out,
    float* __restrict__ scal, float* __restrict__ vec) {
  int node = blockIdx.x;      // b*N + n
  int b    = node >> 7;
  int t    = threadIdx.x;     // feature index

  int sp = nf[node] - 1;      // species 0..4
  float acc = b_embed[t] + W_embed[sp * FF + t];
  for (int tt = 0; tt < 32; ++tt)
    acc += gf[b * 32 + tt] * W_embed[(5 + tt) * FF + t];
  scal[(size_t)node * FF + t] = acc;

#pragma unroll
  for (int m = 0; m < 3; ++m) vec[((size_t)node * FF + t) * 3 + m] = 0.f;
  if (t < 3) pos_out[node * 3 + t] = pos[node * 3 + t];
}

// ---------------------------------------------------------------------------
// K1: z = scal @ Wnode  —  (B*N=512,128) @ (128,128), f16 WMMA, f32 accum
// grid = (32, 8) tiles, 32 threads (one wave) per tile; B pre-swizzled f16
// ---------------------------------------------------------------------------
__global__ __launch_bounds__(32) void node_linear_kernel(
    const float* __restrict__ scal, const _Float16* __restrict__ Wnh,
    float* __restrict__ z) {
  int tm = blockIdx.x;        // 16-row tile over B*N
  int tn = blockIdx.y;        // 16-col tile over F
  int lane = threadIdx.x;
  v8f cacc = {};
#pragma unroll
  for (int kc = 0; kc < 4; ++kc) {
    v16h av = wmma_load_a(scal + (size_t)tm * 16 * FF + kc * 32, FF, lane);
    v16h bv = *(const v16h*)(Wnh + (((size_t)kc * 8 + tn) * 32 + lane) * 16);
    cacc = wmma_f16f32(av, bv, cacc);
  }
  int n = lane & 15, hi = (lane >> 4) & 1;
#pragma unroll
  for (int i = 0; i < 8; ++i)
    z[((size_t)tm * 16 + i + 8 * hi) * FF + tn * 16 + n] = cacc[i];
}

// ---------------------------------------------------------------------------
// K2: per-(batch, receiver) fused message/update kernel. 256 threads = 8 waves.
// Gathers from all 127 senders in 8 chunks of 16 edges; radial MLP second
// matmul (16x64 @ 64x512) runs on v_wmma_f32_16x16x32_f16 with the cutoff
// envelope fused into the D store. h1 is written into LDS already in the
// swizzled f16 A layout so each lane's operand is one contiguous 32B load.
// env comes from the ORIGINAL positions (reference computes the cutoff once
// from lengths_0); env=0 also masks the s==r self edge.
// ---------------------------------------------------------------------------
struct MsgSmem {
  alignas(32) _Float16 h1h[2][32][16];  // swizzled A operand (2 K-chunks)
  float pos[NN][3];
  float posO[NN][3];
  float lng[16];
  float env[16];
  float u[16][3];
  float Y[16][16];
  float fmsg[16][FF];
  union {
    float Rw[16][512];               // radial outputs (per chunk)
    struct {                         // post-processing reuses the same LDS
      float A[16][FF];
      float g[4][FF];
      float scaln[FF];
      float vecn[FF][3];
      float gg[16];
      float red[FF * 3];
    } post;
  } u2;
};

__global__ __launch_bounds__(256) void message_kernel(
    const float* __restrict__ pos0,   // original positions (env)
    const float* __restrict__ pos_in, // current positions
    const float* __restrict__ zbuf,   // (B,N,F)  scal @ Wnode
    const float* __restrict__ vec_in, // (B,N,F,3)
    const float* __restrict__ Wr1, const float* __restrict__ br1,
    const _Float16* __restrict__ Wr2s,// swizzled (2,32,32,16) layer slice
    const float* __restrict__ c1, const float* __restrict__ c2,
    const float* __restrict__ c3,     // (4,F) layer slices
    const float* __restrict__ Wmix0, const float* __restrict__ Wmix1,
    const float* __restrict__ Wg1, const float* __restrict__ Wg2,
    const float* __restrict__ Wvout,
    float* __restrict__ scal_out, float* __restrict__ vec_out,
    float* __restrict__ pos_out) {
  __shared__ MsgSmem sm;

  const int t = threadIdx.x;
  const int b = blockIdx.x >> 7;
  const int r = blockIdx.x & (NN - 1);
  const int lane = t & 31;
  const int wave = t >> 5;

  for (int idx = t; idx < NN * 3; idx += 256) {
    (&sm.pos[0][0])[idx]  = pos_in[(size_t)b * NN * 3 + idx];
    (&sm.posO[0][0])[idx] = pos0[(size_t)b * NN * 3 + idx];
  }
  __syncthreads();

  const float prx = sm.pos[r][0],  pry = sm.pos[r][1],  prz = sm.pos[r][2];
  const float orx = sm.posO[r][0], ory = sm.posO[r][1], orz = sm.posO[r][2];

  float acc[8];
#pragma unroll
  for (int j = 0; j < 8; ++j) acc[j] = 0.f;

  const float* zb = zbuf + (size_t)b * NN * FF;
  const float* vb = vec_in + (size_t)b * NN * FF * 3;

  for (int ch = 0; ch < 8; ++ch) {
    // --- phase 1: edge geometry -------------------------------------------
    if (t < 16) {
      int s = ch * 16 + t;
      float vx = prx - sm.pos[s][0];
      float vy = pry - sm.pos[s][1];
      float vz = prz - sm.pos[s][2];
      float lng = sqrtf(vx * vx + vy * vy + vz * vz + 1e-12f);
      sm.lng[t] = lng;
      float inv = 1.f / lng;
      sm.u[t][0] = vx * inv; sm.u[t][1] = vy * inv; sm.u[t][2] = vz * inv;
      float wx = orx - sm.posO[s][0];
      float wy = ory - sm.posO[s][1];
      float wz = orz - sm.posO[s][2];
      float l0 = sqrtf(wx * wx + wy * wy + wz * wz + 1e-12f);
      float env = 0.f;
      if (s != r && l0 < RMAXF)
        env = 0.5f * (cosf(3.14159265358979f * l0 / RMAXF) + 1.f);
      sm.env[t] = env;
    }
    __syncthreads();
    {   // SH components: 16 edges x 16 comps = 256 = blockDim
      int e = t >> 4, k = t & 15;
      sm.Y[e][k] = sh_comp(k, sm.u[e][0], sm.u[e][1], sm.u[e][2]);
    }
#pragma unroll
    for (int j = 0; j < 4; ++j) {   // h1 = silu(lng*Wr1 + br1): 16x64,
      int idx = t + 256 * j;        // stored swizzled f16 (WMMA A layout)
      int e = idx >> 6, jj = idx & 63;
      float val = silu_f(sm.lng[e] * Wr1[jj] + br1[jj]);
      int kc = jj >> 5, k2 = jj & 31;
      int hi = (k2 >> 3) & 1;
      int jsub = (k2 & 7) + ((k2 >> 4) << 3);
      sm.h1h[kc][e + 16 * hi][jsub] = (_Float16)val;
    }
#pragma unroll
    for (int j = 0; j < 8; ++j) {   // f = z[s] + vec[s].u : 16x128
      int idx = t + 256 * j;
      int e = idx >> 7, f = idx & 127;
      int s = ch * 16 + e;
      const float* vp = vb + ((size_t)s * FF + f) * 3;
      float vd = vp[0] * sm.u[e][0] + vp[1] * sm.u[e][1] + vp[2] * sm.u[e][2];
      sm.fmsg[e][f] = zb[(size_t)s * FF + f] + vd;
    }
    __syncthreads();
    // --- phase 2: radial GEMM 16x64 @ 64x512 on WMMA ----------------------
    v16h av0 = *(const v16h*)(&sm.h1h[0][lane][0]);
    v16h av1 = *(const v16h*)(&sm.h1h[1][lane][0]);
#pragma unroll
    for (int tt = 0; tt < 4; ++tt) {
      int nt = wave * 4 + tt;
      v8f cacc = {};
      v16h bv0 = *(const v16h*)(Wr2s + (((size_t)0 * 32 + nt) * 32 + lane) * 16);
      v16h bv1 = *(const v16h*)(Wr2s + (((size_t)1 * 32 + nt) * 32 + lane) * 16);
      cacc = wmma_f16f32(av0, bv0, cacc);
      cacc = wmma_f16f32(av1, bv1, cacc);
      int n = lane & 15, hi = (lane >> 4) & 1;
#pragma unroll
      for (int i2 = 0; i2 < 8; ++i2) {
        int e = i2 + 8 * hi;                 // D row = edge index
        sm.u2.Rw[e][nt * 16 + n] = cacc[i2] * sm.env[e];
      }
    }
    __syncthreads();
    // --- phase 3: accumulate tensor-product messages ----------------------
#pragma unroll
    for (int j = 0; j < 8; ++j) {
      int idx = t + 256 * j;
      int k = idx >> 7, f = idx & 127;
      int l4 = ell_of(k);
      float a = 0.f;
#pragma unroll
      for (int e = 0; e < 16; ++e)
        a += sm.u2.Rw[e][l4 * FF + f] * sm.Y[e][k] * sm.fmsg[e][f];
      acc[j] += a;
    }
    __syncthreads();   // before next chunk overwrites LDS
  }

  // --- post: A = acc/AVG_NEI, correlation-3 gate, mixes, readout ----------
#pragma unroll
  for (int j = 0; j < 8; ++j) {
    int idx = t + 256 * j;
    int k = idx >> 7, f = idx & 127;
    sm.u2.post.A[k][f] = acc[j] * (1.f / 16.f);
  }
  __syncthreads();
#pragma unroll
  for (int j = 0; j < 2; ++j) {   // per-ell invariants + gate g
    int idx = t + 256 * j;
    int l4 = idx >> 7, f = idx & 127;
    float nrm = 0.f;
#pragma unroll
    for (int k = 0; k < 16; ++k)
      if (ell_of(k) == l4) nrm += sm.u2.post.A[k][f] * sm.u2.post.A[k][f];
    float n2 = nrm * nrm;
    sm.u2.post.g[l4][f] = 1.f + c1[l4 * FF + f] * nrm + c2[l4 * FF + f] * n2 +
                          c3[l4 * FF + f] * n2 * nrm;
  }
  __syncthreads();
#pragma unroll
  for (int j = 0; j < 8; ++j) {
    int idx = t + 256 * j;
    int k = idx >> 7, f = idx & 127;
    sm.u2.post.A[k][f] *= sm.u2.post.g[ell_of(k)][f];
  }
  __syncthreads();
#pragma unroll
  for (int j = 0; j < 2; ++j) {   // 128 scal outputs + 384 vec outputs
    int o = t + 256 * j;
    if (o < FF) {
      float a2 = 0.f;
      for (int cc = 0; cc < FF; ++cc)
        a2 += sm.u2.post.A[0][cc] * Wmix0[cc * FF + o];
      sm.u2.post.scaln[o] = a2;
    } else {
      int q = o - FF;
      int m = q >> 7, fo = q & 127;
      float a2 = 0.f;
      for (int cc = 0; cc < FF; ++cc)
        a2 += sm.u2.post.A[1 + m][cc] * Wmix1[cc * FF + fo];
      sm.u2.post.vecn[fo][m] = a2;
    }
  }
  __syncthreads();
  if (t < 16) {                   // gate hidden: silu(scal @ Wg1)
    float x = 0.f;
    for (int f2 = 0; f2 < FF; ++f2) x += sm.u2.post.scaln[f2] * Wg1[f2 * 16 + t];
    sm.u2.post.gg[t] = silu_f(x);
  }
  __syncthreads();
  if (t < FF) {                   // gate out, weighted vec, store new state
    float gf = 0.f;
#pragma unroll
    for (int g2 = 0; g2 < 16; ++g2) gf += sm.u2.post.gg[g2] * Wg2[g2 * FF + t];
    gf *= Wvout[t];
    sm.u2.post.red[t * 3 + 0] = sm.u2.post.vecn[t][0] * gf;
    sm.u2.post.red[t * 3 + 1] = sm.u2.post.vecn[t][1] * gf;
    sm.u2.post.red[t * 3 + 2] = sm.u2.post.vecn[t][2] * gf;
    size_t base = ((size_t)b * NN + r) * FF + t;
    scal_out[base] = sm.u2.post.scaln[t];
    vec_out[base * 3 + 0] = sm.u2.post.vecn[t][0];
    vec_out[base * 3 + 1] = sm.u2.post.vecn[t][1];
    vec_out[base * 3 + 2] = sm.u2.post.vecn[t][2];
  }
  __syncthreads();
  if (t < 3) {                    // equivariant position update
    float mbv = 0.f;
    for (int f2 = 0; f2 < FF; ++f2) mbv += sm.u2.post.red[f2 * 3 + t];
    pos_out[((size_t)b * NN + r) * 3 + t] = sm.pos[r][t] + mbv;
  }
}

// ---------------------------------------------------------------------------
// K3: out = pos_final - pos0
// ---------------------------------------------------------------------------
__global__ void diff_kernel(const float* __restrict__ p,
                            const float* __restrict__ p0,
                            float* __restrict__ out, int n) {
  int i = blockIdx.x * blockDim.x + threadIdx.x;
  if (i < n) out[i] = p[i] - p0[i];
}

// ---------------------------------------------------------------------------
// host
// ---------------------------------------------------------------------------
extern "C" void kernel_launch(void* const* d_in, const int* in_sizes, int n_in,
                              void* d_out, int out_size, void* d_ws,
                              size_t ws_size, hipStream_t stream) {
  (void)in_sizes; (void)n_in; (void)out_size; (void)ws_size;
  const float* positions       = (const float*)d_in[0];
  const int*   node_features   = (const int*)d_in[1];
  const float* global_features = (const float*)d_in[2];
  const float* W_embed = (const float*)d_in[3];
  const float* b_embed = (const float*)d_in[4];
  const float* Wr1   = (const float*)d_in[5];
  const float* br1   = (const float*)d_in[6];
  const float* Wr2   = (const float*)d_in[7];
  const float* Wnode = (const float*)d_in[8];
  const float* c1    = (const float*)d_in[9];
  const float* c2    = (const float*)d_in[10];
  const float* c3    = (const float*)d_in[11];
  const float* Wmix0 = (const float*)d_in[12];
  const float* Wmix1 = (const float*)d_in[13];
  const float* Wg1   = (const float*)d_in[14];
  const float* Wg2   = (const float*)d_in[15];
  const float* Wvout = (const float*)d_in[16];

  float* ws    = (float*)d_ws;
  float* posA  = ws;                           // B*N*3
  float* posB  = posA + (size_t)BB * NN * 3;
  float* scalA = posB + (size_t)BB * NN * 3;   // B*N*F
  float* scalB = scalA + (size_t)BB * NN * FF;
  float* vecA  = scalB + (size_t)BB * NN * FF; // B*N*F*3
  float* vecB  = vecA + (size_t)BB * NN * FF * 3;
  float* zbuf  = vecB + (size_t)BB * NN * FF * 3;  // B*N*F
  _Float16* wr2h   = (_Float16*)(zbuf + (size_t)BB * NN * FF);  // L*64*512
  _Float16* wnodeh = wr2h + (size_t)LLAYERS * 64 * 512;         // L*128*128

  // one-shot weight swizzles (deterministic; inputs never change)
  swizzle_wr2_kernel<<<(LLAYERS * 64 * 512) / 256, 256, 0, stream>>>(Wr2, wr2h);
  swizzle_wnode_kernel<<<(LLAYERS * FF * FF) / 256, 256, 0, stream>>>(Wnode,
                                                                     wnodeh);

  embed_kernel<<<BB * NN, 128, 0, stream>>>(positions, node_features,
                                            global_features, W_embed, b_embed,
                                            posA, scalA, vecA);

  float* pin = posA;  float* pout = posB;
  float* sin_ = scalA; float* sout = scalB;
  float* vin = vecA;  float* vout = vecB;
  for (int i = 0; i < LLAYERS; ++i) {
    node_linear_kernel<<<dim3(32, 8), 32, 0, stream>>>(
        sin_, wnodeh + (size_t)i * FF * FF, zbuf);
    message_kernel<<<BB * NN, 256, 0, stream>>>(
        positions, pin, zbuf, vin, Wr1 + i * 64, br1 + i * 64,
        wr2h + (size_t)i * 64 * 512, c1 + i * 4 * FF, c2 + i * 4 * FF,
        c3 + i * 4 * FF, Wmix0 + (size_t)i * FF * FF,
        Wmix1 + (size_t)i * FF * FF, Wg1 + (size_t)i * FF * 16,
        Wg2 + (size_t)i * 16 * FF, Wvout + (size_t)i * FF, sout, vout, pout);
    float* tp;
    tp = pin;  pin = pout;  pout = tp;
    tp = sin_; sin_ = sout; sout = tp;
    tp = vin;  vin = vout;  vout = tp;
  }

  diff_kernel<<<(BB * NN * 3 + 255) / 256, 256, 0, stream>>>(
      pin, positions, (float*)d_out, BB * NN * 3);
}
